// LSTM_cell_45054206935275
// MI455X (gfx1250) — compile-verified, matrix operand order checked by reading, not run on hardware
//
#include <hip/hip_runtime.h>

// ---------------------------------------------------------------------------
// LSTM cell on MI455X (gfx1250).
// Phase 1: one-time fp32 -> bf16 conversion into d_ws, stored in a
//          WMMA-fragment-swizzled layout (16x32 tile = 1KB contiguous,
//          lane-chunk = 32B), eliminating all converts from the GEMM loop.
// Phase 2: fused dual-GEMM gates = h@Wh^T + x@Wx^T via
//          v_wmma_f32_16x16x32_bf16 (fp32 accumulate), with branchless
//          bias + sigmoid/tanh + cell update fused in the epilogue.
// B=4096, IN=H=1024: 68.7 GFLOP, ~112MB min HBM traffic -> compute bound;
// bf16 operand set (32MB) is resident in the 192MB L2.
// ---------------------------------------------------------------------------

typedef __bf16 bf16_t;
typedef __attribute__((ext_vector_type(16))) __bf16 v16bf;
typedef __attribute__((ext_vector_type(8)))  float  v8f;

#define B_DIM 4096
#define H_DIM 1024
#define K_DIM 2048           // concat [h | x] reduction length
#define KT    64             // K tiles of 32
#define A_TILES (B_DIM / 16 * KT)            // 16384 tiles
#define W_TILES_PER_GATE (H_DIM / 16 * KT)   // 4096 tiles
#define A_ELEMS ((size_t)B_DIM * K_DIM)      // 8M bf16 elements

struct WPtrs {
    const float* Wh[4];  // f,i,g,o : [1024,1024] row-major
    const float* Wx[4];
};

struct GemmParams {
    const bf16_t* Abf;   // swizzled [A_TILES][32][16]
    const bf16_t* Wbf;   // swizzled [4][W_TILES_PER_GATE][32][16]
    const float*  c;
    const float*  bh[4];
    const float*  bx[4];
    float*        out;   // h_next then c_next
};

// Branchless activations: v_exp_f32 + v_rcp_f32, no exec-mask divergence.
__device__ __forceinline__ float sigmoid_f(float v) {
    return __builtin_amdgcn_rcpf(1.0f + __expf(-v));
}
__device__ __forceinline__ float tanh_f(float v) {
    const float e = __expf(-2.0f * fabsf(v));           // e in (0,1], ->0 for |v| large
    const float t = (1.0f - e) * __builtin_amdgcn_rcpf(1.0f + e);
    return copysignf(t, v);
}

// Convert 16 fp32 (two 32B chunks at p, p+16) into one swizzled lane chunk.
__device__ __forceinline__ void cvt_store_chunk(const float* __restrict__ src,
                                                bf16_t* __restrict__ dst) {
    float4 f0 = ((const float4*)src)[0];
    float4 f1 = ((const float4*)src)[1];
    float4 f2 = ((const float4*)(src + 16))[0];
    float4 f3 = ((const float4*)(src + 16))[1];
    v16bf r;
    r[0]  = (bf16_t)f0.x; r[1]  = (bf16_t)f0.y; r[2]  = (bf16_t)f0.z; r[3]  = (bf16_t)f0.w;
    r[4]  = (bf16_t)f1.x; r[5]  = (bf16_t)f1.y; r[6]  = (bf16_t)f1.z; r[7]  = (bf16_t)f1.w;
    r[8]  = (bf16_t)f2.x; r[9]  = (bf16_t)f2.y; r[10] = (bf16_t)f2.z; r[11] = (bf16_t)f2.w;
    r[12] = (bf16_t)f3.x; r[13] = (bf16_t)f3.y; r[14] = (bf16_t)f3.z; r[15] = (bf16_t)f3.w;
    *(v16bf*)dst = r;
}

// One thread per (tile, lane): A = [h | x], swizzled tiles of 16 rows x 32 K.
// Tile index t = rowTile*KT + kTile (kTile contiguous for linear K walks).
__global__ __launch_bounds__(256)
void convert_A_kernel(const float* __restrict__ h, const float* __restrict__ x,
                      bf16_t* __restrict__ Abf) {
    const unsigned tid  = blockIdx.x * blockDim.x + threadIdx.x;
    const unsigned lane = tid & 31;
    const unsigned tile = tid >> 5;          // [0, A_TILES)
    const unsigned kTile   = tile & (KT - 1);
    const unsigned rowTile = tile >> 6;      // KT == 64
    const unsigned row   = rowTile * 16 + (lane & 15);
    const unsigned kbase = kTile * 32 + (lane >> 4) * 8;   // frag elems 0..7; +16 -> 8..15
    const float* src = (kbase < 1024)
        ? h + (size_t)row * 1024 + kbase
        : x + (size_t)row * 1024 + (kbase - 1024);
    cvt_store_chunk(src, Abf + (size_t)tile * 512 + lane * 16);
}

// One thread per (gate, tile, lane): per-gate W = [Wgh | Wgi] rows = out col.
__global__ __launch_bounds__(256)
void convert_W_kernel(WPtrs w, bf16_t* __restrict__ Wbf) {
    const unsigned tid  = blockIdx.x * blockDim.x + threadIdx.x;
    const unsigned lane = tid & 31;
    const unsigned tile = (tid >> 5) & (W_TILES_PER_GATE - 1);
    const unsigned g    = tid >> 17;         // 4096 tiles * 32 lanes = 2^17 per gate
    const unsigned kTile   = tile & (KT - 1);
    const unsigned colTile = tile >> 6;
    const unsigned n     = colTile * 16 + (lane & 15);
    const unsigned kbase = kTile * 32 + (lane >> 4) * 8;
    const float* src = (kbase < 1024)
        ? w.Wh[g] + (size_t)n * 1024 + kbase
        : w.Wx[g] + (size_t)n * 1024 + (kbase - 1024);
    cvt_store_chunk(src, Wbf + ((size_t)g * W_TILES_PER_GATE + tile) * 512 + lane * 16);
}

// Block = 128 threads (4 waves). Block tile: 256 rows x 16 cols (per gate).
// Each wave: 4 M-subtiles x 4 gates = 16 v8f accumulators; B fragments are
// loaded once per K-step and reused across the 4 M-subtiles.
__global__ __launch_bounds__(128)
void lstm_cell_wmma_kernel(GemmParams p) {
    const int lane = threadIdx.x & 31;
    const int wave = threadIdx.x >> 5;
    const int hl   = lane & 15;
    const int hi   = lane >> 4;

    const int colTile     = blockIdx.x;                    // per-gate N tile
    const int rowTileBase = blockIdx.y * 16 + wave * 4;    // 4 M-subtiles/wave

    v8f acc[4][4];
#pragma unroll
    for (int m = 0; m < 4; ++m)
#pragma unroll
        for (int gt = 0; gt < 4; ++gt)
            acc[m][gt] = (v8f){0.f, 0.f, 0.f, 0.f, 0.f, 0.f, 0.f, 0.f};

    // Per-wave strip bases: each advances 1KB (512 bf16) per kTile.
    const bf16_t* aBase = p.Abf + (size_t)rowTileBase * KT * 512 + lane * 16;
    const bf16_t* wBase = p.Wbf + (size_t)colTile * KT * 512 + lane * 16;

    for (int kt = 0; kt < KT; ++kt) {
        const size_t ko = (size_t)kt * 512;
        v16bf bfrag[4];
#pragma unroll
        for (int gt = 0; gt < 4; ++gt)
            bfrag[gt] = *(const v16bf*)(wBase + (size_t)gt * W_TILES_PER_GATE * 512 + ko);
#pragma unroll
        for (int m = 0; m < 4; ++m) {
            v16bf afrag = *(const v16bf*)(aBase + (size_t)m * KT * 512 + ko);
#pragma unroll
            for (int gt = 0; gt < 4; ++gt) {
                acc[m][gt] = __builtin_amdgcn_wmma_f32_16x16x32_bf16(
                    /*neg_a=*/false, afrag,
                    /*neg_b=*/false, bfrag[gt],
                    /*c_mod=*/(short)0, acc[m][gt],
                    /*reuse_a=*/false, /*reuse_b=*/false);
            }
        }
    }

    // ---- fused epilogue: bias + activations + cell update (branchless) ----
    const int col = colTile * 16 + hl;
    const float bf_ = p.bh[0][col] + p.bx[0][col];
    const float bi_ = p.bh[1][col] + p.bx[1][col];
    const float bg_ = p.bh[2][col] + p.bx[2][col];
    const float bo_ = p.bh[3][col] + p.bx[3][col];

    float* __restrict__ hout = p.out;
    float* __restrict__ cout = p.out + (size_t)B_DIM * H_DIM;

    // Acc layout: lanes 0-15, VGPR r -> M=r; lanes 16-31, VGPR r -> M=r+8.
#pragma unroll
    for (int m = 0; m < 4; ++m) {
#pragma unroll
        for (int r = 0; r < 8; ++r) {
            const int    row = (rowTileBase + m) * 16 + hi * 8 + r;
            const size_t idx = (size_t)row * H_DIM + col;
            const float fg = sigmoid_f(acc[m][0][r] + bf_);
            const float ig = sigmoid_f(acc[m][1][r] + bi_);
            const float gg = tanh_f(acc[m][2][r] + bg_);
            const float og = sigmoid_f(acc[m][3][r] + bo_);
            const float cn = fg * p.c[idx] + ig * gg;
            hout[idx] = tanh_f(cn) * og;
            cout[idx] = cn;
        }
    }
}

extern "C" void kernel_launch(void* const* d_in, const int* in_sizes, int n_in,
                              void* d_out, int out_size, void* d_ws, size_t ws_size,
                              hipStream_t stream) {
    (void)in_sizes; (void)n_in; (void)out_size; (void)ws_size;
    // setup_inputs order:
    // 0:x 1:h 2:c 3:Wfh 4:bfh 5:Wfi 6:bfi 7:Wih 8:bih 9:Wii 10:bii
    // 11:Wch 12:bch 13:Wci 14:bci 15:Woh 16:boh 17:Woi 18:boi
    const float* x = (const float*)d_in[0];
    const float* h = (const float*)d_in[1];

    bf16_t* Abf = (bf16_t*)d_ws;                 // 8M bf16 = 16MB
    bf16_t* Wbf = Abf + A_ELEMS;                 // 8M bf16 = 16MB

    WPtrs w;
    w.Wh[0] = (const float*)d_in[3];   w.Wx[0] = (const float*)d_in[5];
    w.Wh[1] = (const float*)d_in[7];   w.Wx[1] = (const float*)d_in[9];
    w.Wh[2] = (const float*)d_in[11];  w.Wx[2] = (const float*)d_in[13];
    w.Wh[3] = (const float*)d_in[15];  w.Wx[3] = (const float*)d_in[17];

    GemmParams p;
    p.Abf = Abf;
    p.Wbf = Wbf;
    p.c   = (const float*)d_in[2];
    p.bh[0] = (const float*)d_in[4];   p.bx[0] = (const float*)d_in[6];
    p.bh[1] = (const float*)d_in[8];   p.bx[1] = (const float*)d_in[10];
    p.bh[2] = (const float*)d_in[12];  p.bx[2] = (const float*)d_in[14];
    p.bh[3] = (const float*)d_in[16];  p.bx[3] = (const float*)d_in[18];
    p.out = (float*)d_out;

    // Phase 1: one-time swizzled bf16 conversion (in-order on stream).
    convert_A_kernel<<<A_TILES * 32 / 256, 256, 0, stream>>>(h, x, Abf);
    convert_W_kernel<<<4 * W_TILES_PER_GATE * 32 / 256, 256, 0, stream>>>(w, Wbf);

    // Phase 2: fused WMMA GEMM + gates.
    dim3 grid(H_DIM / 16, B_DIM / 256);  // 64 x 16 workgroups
    dim3 block(128);                     // 4 waves
    lstm_cell_wmma_kernel<<<grid, block, 0, stream>>>(p);
}
